// GNNRetriever_22969485099231
// MI455X (gfx1250) — matrix operand order repaired and checked by reading, not compiled, not run on hardware
//
#include <hip/hip_runtime.h>
#include <stdint.h>

#define N_NODES 50000
#define N_EDGES 800000
#define NODE_DIM 256
#define EDGE_DIM 64
#define HIDDEN 256
#define Q_DIM 768
#define NEG_SLOPE 0.2f

typedef __attribute__((ext_vector_type(16))) __bf16 v16bf;
typedef __attribute__((ext_vector_type(8)))  __bf16 v8bf;
typedef __attribute__((ext_vector_type(8)))  float  v8f;

__device__ __forceinline__ unsigned short f2bf(float f) {
    unsigned u = __float_as_uint(f);
    unsigned r = u + 0x7fffu + ((u >> 16) & 1u);   // round-to-nearest-even
    return (unsigned short)(r >> 16);
}

// ---------------- conversion / packing kernels ----------------

__global__ void k_f32_to_bf16(const float* __restrict__ in,
                              unsigned short* __restrict__ out, int n) {
    int i = blockIdx.x * blockDim.x + threadIdx.x;
    if (i < n) out[i] = f2bf(in[i]);
}

// Wt[n][k] = W[k][n]  (bf16), W is [kdim x 256] row-major
__global__ void k_w_transpose_bf16(const float* __restrict__ W,
                                   unsigned short* __restrict__ Wt, int kdim) {
    int t = blockIdx.x * blockDim.x + threadIdx.x;
    if (t >= kdim * 256) return;
    int k = t >> 8, n = t & 255;
    Wt[n * kdim + k] = f2bf(W[k * 256 + n]);
}

// ---------------- WMMA GEMM: C[N x 256] = A(bf16) x B(bf16, col-major) ----------------
// One wave computes one 16x16 C tile; K looped 8 x 32 with v_wmma_f32_16x16x32_bf16.
__global__ void __launch_bounds__(256)
k_gemm_bf16_wmma(const unsigned short* __restrict__ A,   // [N][256] bf16 row-major
                 const unsigned short* __restrict__ Bt,  // [256][256] bf16, Bt[n][k]
                 float* __restrict__ C,                   // [N][256] f32
                 const float* __restrict__ addvec,        // [256] or null
                 int do_relu) {
    int lane = threadIdx.x & 31;
    int wave = threadIdx.x >> 5;
    int gw   = blockIdx.x * 8 + wave;      // 50000 tiles total
    int tileM = gw >> 4;                    // 0..3124
    int tileN = gw & 15;                    // 0..15

    const __bf16* Ab = (const __bf16*)A;
    const __bf16* Bb = (const __bf16*)Bt;

    // A 16x32 bf16 fragment: lane m=lane&15, kbase = lane<16 ? 0 : 16,
    //             halves [kbase..kbase+7] and [kbase+8..kbase+15]
    int mrow = tileM * 16 + (lane & 15);
    int akb  = (lane < 16) ? 0 : 16;
    // B 32x16: lane n=lane&15, 16 contiguous K-halves at kbase2 = lane<16 ? 0 : 16
    int ncol = tileN * 16 + (lane & 15);
    int bkb  = (lane < 16) ? 0 : 16;

    const __bf16* pa = Ab + (size_t)mrow * 256;
    const __bf16* pb = Bb + (size_t)ncol * 256;

    v8f c = {};
#pragma unroll
    for (int kk = 0; kk < 256; kk += 32) {
        union { v16bf v; v8bf h[2]; } af;
        af.h[0] = *(const v8bf*)(pa + kk + akb);
        af.h[1] = *(const v8bf*)(pa + kk + akb + 8);
        v16bf bfr = *(const v16bf*)(pb + kk + bkb);
        c = __builtin_amdgcn_wmma_f32_16x16x32_bf16(
                false, af.v, false, bfr, (short)0, c, false, false);
    }

    // C/D layout: lane n = lane&15; VGPR r -> row r (lanes 0-15) / 8+r (lanes 16-31)
    int n = tileN * 16 + (lane & 15);
    int mbase = tileM * 16 + ((lane < 16) ? 0 : 8);
    float av = addvec ? addvec[n] : 0.0f;
#pragma unroll
    for (int r = 0; r < 8; ++r) {
        float v = c[r] + av;
        if (do_relu) v = fmaxf(v, 0.0f);
        C[(size_t)(mbase + r) * 256 + n] = v;
    }
}

// ---------------- per-layer init ----------------
__global__ void k_layer_init(float* __restrict__ AGG, unsigned* __restrict__ amaxU,
                             float* __restrict__ denom, float* __restrict__ aesum) {
    int i = blockIdx.x * blockDim.x + threadIdx.x;
    if (i < N_NODES * 256) AGG[i] = 0.0f;
    if (i < N_NODES) { amaxU[i] = 0u; denom[i] = 1e-16f; }
    if (i == 0) aesum[0] = 0.0f;
}

// ve[r] = dot(We[r][:256], ae)   (We: [64][256])
__global__ void k_ve(const float* __restrict__ We, const float* __restrict__ ae,
                     float* __restrict__ ve) {
    int lane = threadIdx.x & 31;
    int row  = (blockIdx.x * blockDim.x + threadIdx.x) >> 5;
    if (row >= EDGE_DIM) return;
    float s = 0.f;
    for (int j = lane; j < 256; j += 32) s += We[row * 256 + j] * ae[j];
    for (int off = 16; off > 0; off >>= 1) s += __shfl_down(s, off, 32);
    if (lane == 0) ve[row] = s;
}

// ae_edges[e] = edge_attr[e][:64] . ve ; also accumulate sum into aesum
__global__ void __launch_bounds__(256)
k_ae_edges(const float* __restrict__ ea, const float* __restrict__ ve,
           float* __restrict__ ae_edges, float* __restrict__ aesum) {
    __shared__ float bsum;
    if (threadIdx.x == 0) bsum = 0.f;
    __syncthreads();
    int lane = threadIdx.x & 31;
    int e = (blockIdx.x * blockDim.x + threadIdx.x) >> 5;
    if (e < N_EDGES) {
        float s = ea[(size_t)e * 64 + lane] * ve[lane]
                + ea[(size_t)e * 64 + 32 + lane] * ve[32 + lane];
        for (int off = 16; off > 0; off >>= 1) s += __shfl_down(s, off, 32);
        if (lane == 0) { ae_edges[e] = s; atomicAdd(&bsum, s); }
    }
    __syncthreads();
    if (threadIdx.x == 0) atomicAdd(aesum, bsum);
}

// s_src[i] = H[i].a_src ; s_dst[i] = H[i].a_dst
__global__ void __launch_bounds__(256)
k_sdots(const float* __restrict__ H, const float* __restrict__ av,
        const float* __restrict__ dv, float* __restrict__ ssrc,
        float* __restrict__ sdst) {
    int lane = threadIdx.x & 31;
    int nidx = (blockIdx.x * blockDim.x + threadIdx.x) >> 5;
    if (nidx >= N_NODES) return;
    const float* hr = H + (size_t)nidx * 256;
    float s1 = 0.f, s2 = 0.f;
    for (int j = lane; j < 256; j += 32) {
        float h = hr[j];
        s1 += h * av[j];
        s2 += h * dv[j];
    }
    for (int off = 16; off > 0; off >>= 1) {
        s1 += __shfl_down(s1, off, 32);
        s2 += __shfl_down(s2, off, 32);
    }
    if (lane == 0) { ssrc[nidx] = s1; sdst[nidx] = s2; }
}

// alpha = leakyrelu(ssrc[s]+sdst[d]+ae) ; segment max via monotone-uint atomicMax
__global__ void k_alpha_max(const int* __restrict__ src, const int* __restrict__ dst,
                            const float* __restrict__ ae_edges,
                            const float* __restrict__ aesum,
                            const float* __restrict__ ssrc, const float* __restrict__ sdst,
                            float* __restrict__ alpha, unsigned* __restrict__ amaxU) {
    int t = blockIdx.x * blockDim.x + threadIdx.x;
    if (t >= N_EDGES + N_NODES) return;
    int s, d; float add;
    if (t < N_EDGES) { s = src[t]; d = dst[t]; add = ae_edges[t]; }
    else { s = d = t - N_EDGES; add = aesum[0] * (1.0f / N_EDGES); }
    float a = ssrc[s] + sdst[d] + add;
    a = (a > 0.f) ? a : NEG_SLOPE * a;
    alpha[t] = a;
    unsigned u = __float_as_uint(a);
    unsigned key = (u & 0x80000000u) ? ~u : (u | 0x80000000u);
    atomicMax(&amaxU[d], key);
}

// ex = exp(alpha - amax[d]) ; denom[d] += ex
__global__ void k_expsum(const int* __restrict__ dst, const unsigned* __restrict__ amaxU,
                         float* __restrict__ alpha, float* __restrict__ denom) {
    int t = blockIdx.x * blockDim.x + threadIdx.x;
    if (t >= N_EDGES + N_NODES) return;
    int d = (t < N_EDGES) ? dst[t] : (t - N_EDGES);
    unsigned k = amaxU[d];
    unsigned u = (k & 0x80000000u) ? (k & 0x7fffffffu) : ~k;
    float m = __uint_as_float(u);
    float ex = __expf(alpha[t] - m);
    alpha[t] = ex;
    atomicAdd(&denom[d], ex);
}

// AGG[d][:] += (ex/denom[d]) * H[s][:]   (wave per edge, 8 feats/lane, coalesced)
__global__ void __launch_bounds__(256)
k_aggregate(const int* __restrict__ src, const int* __restrict__ dst,
            const float* __restrict__ alpha, const float* __restrict__ denom,
            const float* __restrict__ H, float* __restrict__ AGG) {
    int lane = threadIdx.x & 31;
    int t = (blockIdx.x * blockDim.x + threadIdx.x) >> 5;
    if (t >= N_EDGES + N_NODES) return;
    int s, d;
    if (t < N_EDGES) { s = src[t]; d = dst[t]; }
    else { s = d = t - N_EDGES; }
    float w = alpha[t] / denom[d];
    const float* hs = H + (size_t)s * 256;
    float* od = AGG + (size_t)d * 256;
#pragma unroll
    for (int j = 0; j < 8; ++j) {
        int c = lane + j * 32;
        atomicAdd(&od[c], w * hs[c]);
    }
}

// next-layer input: Xb = bf16(relu(AGG + b))
__global__ void k_bias_relu_to_bf16(const float* __restrict__ AGG,
                                    const float* __restrict__ b,
                                    unsigned short* __restrict__ Xb) {
    int i = blockIdx.x * blockDim.x + threadIdx.x;
    if (i >= N_NODES * 256) return;
    float v = AGG[i] + b[i & 255];
    Xb[i] = f2bf(fmaxf(v, 0.f));
}

// qc[j] = bs1[j] + sum_k q[k] * Ws1[(256+k)*256 + j]
__global__ void __launch_bounds__(256)
k_qconst(const float* __restrict__ q, const float* __restrict__ Ws1,
         const float* __restrict__ bs1, float* __restrict__ qc) {
    int lane = threadIdx.x & 31;
    int j = (blockIdx.x * blockDim.x + threadIdx.x) >> 5;
    if (j >= 256) return;
    float s = 0.f;
    for (int k = lane; k < Q_DIM; k += 32) s += q[k] * Ws1[(size_t)(256 + k) * 256 + j];
    for (int off = 16; off > 0; off >>= 1) s += __shfl_down(s, off, 32);
    if (lane == 0) qc[j] = s + bs1[j];
}

// logits[n] = S[n] . Ws2 + bs2
__global__ void __launch_bounds__(256)
k_score(const float* __restrict__ S, const float* __restrict__ Ws2,
        const float* __restrict__ bs2, float* __restrict__ out) {
    int lane = threadIdx.x & 31;
    int nidx = (blockIdx.x * blockDim.x + threadIdx.x) >> 5;
    if (nidx >= N_NODES) return;
    float s = 0.f;
    for (int j = lane; j < 256; j += 32) s += S[(size_t)nidx * 256 + j] * Ws2[j];
    for (int off = 16; off > 0; off >>= 1) s += __shfl_down(s, off, 32);
    if (lane == 0) out[nidx] = s + bs2[0];
}

// ---------------- host orchestration ----------------

extern "C" void kernel_launch(void* const* d_in, const int* in_sizes, int n_in,
                              void* d_out, int out_size, void* d_ws, size_t ws_size,
                              hipStream_t stream) {
    (void)in_sizes; (void)n_in; (void)out_size; (void)ws_size;
    const float* x   = (const float*)d_in[0];
    const int*   ei  = (const int*)d_in[1];
    const float* ea  = (const float*)d_in[2];
    const float* q   = (const float*)d_in[3];
    const float* W1  = (const float*)d_in[4];
    const float* as1 = (const float*)d_in[5];
    const float* ad1 = (const float*)d_in[6];
    const float* We1 = (const float*)d_in[7];
    const float* ae1 = (const float*)d_in[8];
    const float* b1  = (const float*)d_in[9];
    const float* W2  = (const float*)d_in[10];
    const float* as2 = (const float*)d_in[11];
    const float* ad2 = (const float*)d_in[12];
    const float* We2 = (const float*)d_in[13];
    const float* ae2 = (const float*)d_in[14];
    const float* b2  = (const float*)d_in[15];
    const float* Ws1 = (const float*)d_in[16];
    const float* bs1 = (const float*)d_in[17];
    const float* Ws2 = (const float*)d_in[18];
    const float* bs2 = (const float*)d_in[19];

    const int* srcp = ei;
    const int* dstp = ei + N_EDGES;

    char* wsp = (char*)d_ws;
    size_t off = 0;
    auto alloc = [&](size_t bytes) -> void* {
        void* p = wsp + off;
        off += (bytes + 255) & ~(size_t)255;
        return p;
    };

    unsigned short* Xb  = (unsigned short*)alloc((size_t)N_NODES * 256 * 2); // bf16 act
    unsigned short* Wt  = (unsigned short*)alloc(256 * 256 * 2);             // bf16 W^T
    float*    G        = (float*)alloc((size_t)N_NODES * 256 * 4);           // gemm out
    float*    AGG      = (float*)alloc((size_t)N_NODES * 256 * 4);           // scatter acc
    float*    ssrc     = (float*)alloc((size_t)N_NODES * 4);
    float*    sdst     = (float*)alloc((size_t)N_NODES * 4);
    float*    ae_edges = (float*)alloc((size_t)N_EDGES * 4);
    float*    alpha    = (float*)alloc((size_t)(N_EDGES + N_NODES) * 4);
    unsigned* amaxU    = (unsigned*)alloc((size_t)N_NODES * 4);
    float*    denom    = (float*)alloc((size_t)N_NODES * 4);
    float*    ve       = (float*)alloc(EDGE_DIM * 4);
    float*    aesum    = (float*)alloc(4);
    float*    qc       = (float*)alloc(256 * 4);

    const int BLK = 256;
    const int gNF   = (N_NODES * 256) / BLK;            // 50000
    const int gTile = (N_NODES / 16) * 16 / 8;          // 6250 (8 waves/block)
    const int gNode = (N_NODES + 7) / 8;                // 6250 (wave/node)
    const int gEN   = (N_EDGES + N_NODES + BLK - 1) / BLK;
    const int gENw  = (N_EDGES + N_NODES + 7) / 8;      // wave per edge+loop
    const int gE    = (N_EDGES + 7) / 8;                // wave per edge

    // x -> bf16
    k_f32_to_bf16<<<gNF, BLK, 0, stream>>>(x, Xb, N_NODES * 256);

    auto gat_layer = [&](const float* W, const float* as_, const float* ad_,
                         const float* We, const float* ae_, const float* b_) {
        k_w_transpose_bf16<<<256, BLK, 0, stream>>>(W, Wt, 256);
        k_gemm_bf16_wmma<<<gTile, BLK, 0, stream>>>(Xb, Wt, G, nullptr, 0);
        k_layer_init<<<gNF, BLK, 0, stream>>>(AGG, amaxU, denom, aesum);
        k_ve<<<8, BLK, 0, stream>>>(We, ae_, ve);
        k_ae_edges<<<gE, BLK, 0, stream>>>(ea, ve, ae_edges, aesum);
        k_sdots<<<gNode, BLK, 0, stream>>>(G, as_, ad_, ssrc, sdst);
        k_alpha_max<<<gEN, BLK, 0, stream>>>(srcp, dstp, ae_edges, aesum,
                                             ssrc, sdst, alpha, amaxU);
        k_expsum<<<gEN, BLK, 0, stream>>>(dstp, amaxU, alpha, denom);
        k_aggregate<<<gENw, BLK, 0, stream>>>(srcp, dstp, alpha, denom, G, AGG);
        k_bias_relu_to_bf16<<<gNF, BLK, 0, stream>>>(AGG, b_, Xb);
    };

    gat_layer(W1, as1, ad1, We1, ae1, b1);   // Xb <- relu(gat1(x))
    gat_layer(W2, as2, ad2, We2, ae2, b2);   // Xb <- relu(gat2(h1))

    // scorer: S = relu(h2 @ Ws1[:256] + (q @ Ws1[256:] + bs1)); logits = S @ Ws2 + bs2
    k_w_transpose_bf16<<<256, BLK, 0, stream>>>(Ws1, Wt, 256);
    k_qconst<<<32, BLK, 0, stream>>>(q, Ws1, bs1, qc);
    k_gemm_bf16_wmma<<<gTile, BLK, 0, stream>>>(Xb, Wt, G, qc, 1);
    k_score<<<gNode, BLK, 0, stream>>>(G, Ws2, bs2, (float*)d_out);
}
